// Attention_Layer_9242769622375
// MI455X (gfx1250) — matrix operand order repaired and checked by reading, not compiled, not run on hardware
//
#include <hip/hip_runtime.h>
#include <hip/hip_bf16.h>

typedef __attribute__((ext_vector_type(16))) _Float16 v16h;
typedef __attribute__((ext_vector_type(8)))  _Float16 v8h;
typedef __attribute__((ext_vector_type(8)))  float    v8f;
typedef __attribute__((ext_vector_type(4)))  float    v4f;

#define T_LEN   200
#define D_DIM   64
#define H1_DIM  80
#define H2_DIM  40
#define MT_TILES 13            // ceil(200/16) -> 208 padded rows
#define H1S_STRIDE 104         // halves per row (96 used + pad, conflict-friendly)
#define KST_STRIDE 68          // floats per staged k row; 272B row stride = 16B aligned
#define NEG_F (-4294967295.0f)

// Async global->LDS copy, 16B per lane, tracked by ASYNCcnt (gfx1250).
// lds_addr is the wave-relative LDS byte offset (low 32 bits of the flat
// aperture address of a __shared__ object, per ISA 10.2).
__device__ __forceinline__ void async_copy_b128(const float* gsrc, void* ldst) {
    unsigned lds_off = (unsigned)(uintptr_t)ldst;
    unsigned long long gaddr = (unsigned long long)(uintptr_t)gsrc;
    __asm__ volatile("global_load_async_to_lds_b128 %0, %1, off"
                     :
                     : "v"(lds_off), "v"(gaddr)
                     : "memory");
}

__device__ __forceinline__ void wait_async_zero() {
    __asm__ volatile("s_wait_asynccnt 0" ::: "memory");
}

// --- weight prep: f32 row-major -> f16 column-major, padded for WMMA B fragments ---
// W1h: [80 cols][256 K]   (exact)
// W2h: [48 cols][96 K]    (zero-padded from 40x80)
__global__ __launch_bounds__(256)
void din_prep_weights(const float* __restrict__ W1, const float* __restrict__ W2,
                      _Float16* __restrict__ W1h, _Float16* __restrict__ W2h) {
    const int stride = gridDim.x * blockDim.x;
    const int tid = blockIdx.x * blockDim.x + threadIdx.x;
    for (int i = tid; i < H1_DIM * 256; i += stride) {
        int n = i >> 8, c = i & 255;
        W1h[i] = (_Float16)W1[c * H1_DIM + n];
    }
    for (int i = tid; i < 48 * 96; i += stride) {
        int n = i / 96, kk = i - n * 96;
        float w = (n < H2_DIM && kk < H1_DIM) ? W2[kk * H2_DIM + n] : 0.0f;
        W2h[i] = (_Float16)w;
    }
}

__global__ __launch_bounds__(128)
void din_attn_kernel(const float* __restrict__ q, const float* __restrict__ kmat,
                     const float* __restrict__ v, const int* __restrict__ mask,
                     const float* __restrict__ b1, const float* __restrict__ b2,
                     const float* __restrict__ Wf, const float* __restrict__ bf,
                     const _Float16* __restrict__ W1h, const _Float16* __restrict__ W2h,
                     float* __restrict__ out) {
    __shared__ _Float16 h1s[208 * H1S_STRIDE];      // 43264 B, f16 h1 (K-padded to 96)
    __shared__ float    ksts[4][16 * KST_STRIDE];   // 17408 B, per-wave k tiles
    __shared__ float    qs[64];
    __shared__ float    logits[208];
    __shared__ float    red[128];

    const int b     = blockIdx.x;
    const int tid   = threadIdx.x;
    const int wave  = tid >> 5;
    const int lane  = tid & 31;
    const int row16 = lane & 15;
    const int hi    = lane >> 4;

    // ---- init: zero h1 (covers K padding), seed logits with bf, stage q ----
    {
        int* h1w = (int*)h1s;
        for (int i = tid; i < (208 * H1S_STRIDE) / 2; i += 128) h1w[i] = 0;
        const float bfv = bf[0];
        for (int t = tid; t < 208; t += 128) logits[t] = bfv;
        if (tid < 64) qs[tid] = q[b * 64 + tid];
    }
    __syncthreads();

    const float* kb = kmat + (size_t)b * T_LEN * D_DIM;
    float* kw = ksts[wave];
    const float* kwr = kw + row16 * KST_STRIDE;

    for (int mt = wave; mt < MT_TILES; mt += 4) {
        const int t0 = mt * 16;

        // ---- stage 16x64 k tile into LDS, async global->LDS (zero-pad beyond T) ----
        for (int i = lane; i < 256; i += 32) {          // 256 chunks of 16B
            int r = i >> 4, c = (i & 15) << 2;
            int t = t0 + r;
            float* dstg = kw + r * KST_STRIDE + c;
            if (t < T_LEN) {
                async_copy_b128(kb + t * 64 + c, dstg);
            } else {
                *(v4f*)dstg = (v4f){0.f, 0.f, 0.f, 0.f};
            }
        }
        wait_async_zero();

        // pull next tile toward caches (global_prefetch_b8)
        if (mt + 4 < MT_TILES) {
            int tn = (mt + 4) * 16 + row16;
            if (tn < T_LEN) __builtin_prefetch(kb + tn * 64 + hi * 32, 0, 1);
        }

        // ---- GEMM1: info[16x256] x W1h -> h1[16x80] ----
#pragma unroll 1
        for (int nt = 0; nt < 5; ++nt) {
            v8f acc = {0.f, 0.f, 0.f, 0.f, 0.f, 0.f, 0.f, 0.f};
#pragma unroll
            for (int ks = 0; ks < 8; ++ks) {
                // synthesize A fragment in registers from q/k in LDS
                v16h a;
#pragma unroll
                for (int j = 0; j < 8; ++j) {
                    int g0 = ks * 32 + hi * 16 + j;       // lane elements 0..7
                    int g1 = ks * 32 + hi * 16 + 8 + j;   // lane elements 8..15
                    int s0 = g0 >> 6, c0 = g0 & 63;
                    int s1 = g1 >> 6, c1 = g1 & 63;
                    float q0 = qs[c0], k0v = kwr[c0];
                    float q1 = qs[c1], k1v = kwr[c1];
                    float f0 = (s0 == 0) ? q0 : (s0 == 1) ? k0v
                             : (s0 == 2) ? (q0 - k0v) : (q0 * k0v);
                    float f1 = (s1 == 0) ? q1 : (s1 == 1) ? k1v
                             : (s1 == 2) ? (q1 - k1v) : (q1 * k1v);
                    a[j]     = (_Float16)f0;
                    a[8 + j] = (_Float16)f1;
                }
                const v16h bw = *(const v16h*)(W1h + ((nt * 16 + row16) << 8)
                                               + ks * 32 + hi * 16);
                acc = __builtin_amdgcn_wmma_f32_16x16x32_f16(
                          false, a, false, bw, (short)0, acc, false, false);
            }
            const int   n    = nt * 16 + row16;
            const float bias = b1[n];
#pragma unroll
            for (int r = 0; r < 8; ++r) {
                float hv = acc[r] + bias;
                hv = hv > 0.f ? hv : 0.f;
                h1s[(t0 + hi * 8 + r) * H1S_STRIDE + n] = (_Float16)hv;
            }
        }

        // ---- GEMM2 + folded Wf projection: h1[16x96] x W2h -> logits ----
#pragma unroll 1
        for (int nt2 = 0; nt2 < 3; ++nt2) {
            v8f acc = {0.f, 0.f, 0.f, 0.f, 0.f, 0.f, 0.f, 0.f};
#pragma unroll
            for (int ks2 = 0; ks2 < 3; ++ks2) {
                const _Float16* arow = h1s + (t0 + row16) * H1S_STRIDE
                                           + ks2 * 32 + hi * 16;
                v8h lo  = *(const v8h*)(arow);
                v8h hi8 = *(const v8h*)(arow + 8);
                v16h a;
#pragma unroll
                for (int j = 0; j < 8; ++j) { a[j] = lo[j]; a[8 + j] = hi8[j]; }
                const v16h bw = *(const v16h*)(W2h + (nt2 * 16 + row16) * 96
                                               + ks2 * 32 + hi * 16);
                acc = __builtin_amdgcn_wmma_f32_16x16x32_f16(
                          false, a, false, bw, (short)0, acc, false, false);
            }
            const int   n   = nt2 * 16 + row16;
            const float wfv = (n < H2_DIM) ? Wf[n] : 0.0f;
            const float b2v = (n < H2_DIM) ? b2[n] : 0.0f;
#pragma unroll
            for (int r = 0; r < 8; ++r) {
                float hv = acc[r] + b2v;
                hv = hv > 0.f ? hv : 0.f;
                float p = hv * wfv;
                // reduce across the 16 lanes sharing the same output row
                p += __shfl_xor(p, 1, 16);
                p += __shfl_xor(p, 2, 16);
                p += __shfl_xor(p, 4, 16);
                p += __shfl_xor(p, 8, 16);
                if (row16 == 0) logits[t0 + hi * 8 + r] += p;  // rows owned by this wave
            }
        }
    }
    __syncthreads();

    // ---- mask + softmax over T ----
    const int* mb = mask + b * T_LEN;
    float lmax = NEG_F;
    for (int t = tid; t < T_LEN; t += 128) {
        float lg = (mb[t] == 0) ? NEG_F : logits[t];
        logits[t] = lg;
        lmax = lmax > lg ? lmax : lg;
    }
    red[tid] = lmax;
    __syncthreads();
    for (int s = 64; s > 0; s >>= 1) {
        if (tid < s) { float o = red[tid + s]; if (o > red[tid]) red[tid] = o; }
        __syncthreads();
    }
    const float m = red[0];
    __syncthreads();
    float lsum = 0.f;
    for (int t = tid; t < T_LEN; t += 128) {
        float e = __expf(logits[t] - m);
        logits[t] = e;
        lsum += e;
    }
    red[tid] = lsum;
    __syncthreads();
    for (int s = 64; s > 0; s >>= 1) {
        if (tid < s) red[tid] += red[tid + s];
        __syncthreads();
    }
    const float inv = 1.0f / red[0];
    __syncthreads();

    // ---- out[b,d] = inv * sum_t e[t] * v[b,t,d]; T split 2-ways ----
    {
        const float* vb = v + (size_t)b * T_LEN * D_DIM;
        int d    = tid & 63;
        int half = tid >> 6;
        int tb   = half * (T_LEN / 2);
        int te   = tb + (T_LEN / 2);
        float accum = 0.f;
        for (int t = tb; t < te; ++t) accum += logits[t] * vb[t * 64 + d];
        red[tid] = accum;
    }
    __syncthreads();
    if (tid < 64) out[b * 64 + tid] = (red[tid] + red[tid + 64]) * inv;
}

extern "C" void kernel_launch(void* const* d_in, const int* in_sizes, int n_in,
                              void* d_out, int out_size, void* d_ws, size_t ws_size,
                              hipStream_t stream) {
    const float* q    = (const float*)d_in[0];
    const float* k    = (const float*)d_in[1];
    const float* v    = (const float*)d_in[2];
    const int*   mask = (const int*)d_in[3];
    const float* W1   = (const float*)d_in[4];
    const float* b1   = (const float*)d_in[5];
    const float* W2   = (const float*)d_in[6];
    const float* b2   = (const float*)d_in[7];
    const float* Wf   = (const float*)d_in[8];
    const float* bf   = (const float*)d_in[9];
    float* out = (float*)d_out;

    const int B = in_sizes[0] / D_DIM;

    _Float16* W1h = (_Float16*)d_ws;           // 80*256 halves
    _Float16* W2h = W1h + H1_DIM * 256;        // 48*96 halves

    din_prep_weights<<<64, 256, 0, stream>>>(W1, W2, W1h, W2h);
    din_attn_kernel<<<B, 128, 0, stream>>>(q, k, v, mask, b1, b2, Wf, bf,
                                           W1h, W2h, out);
}
